// Fastfood_2dConv_65515431133407
// MI455X (gfx1250) — matrix operand order repaired and checked by reading, not compiled, not run on hardware
//
#include <hip/hip_runtime.h>
#include <hip/hip_bf16.h>

typedef __attribute__((ext_vector_type(16))) _Float16 v16h;
typedef __attribute__((ext_vector_type(8)))  float    v8f;

#define LDS_PITCH 65   // 64 + 1 pad to break LDS bank conflicts on 2D access

__device__ __forceinline__ _Float16 hsign(int r, int c) {
    return (_Float16)((__popc(r & c) & 1) ? -1.0f : 1.0f);
}

// dst = H64 * src   (64x64 f32 tiles in LDS, pitch LDS_PITCH)
// Wave `wave` computes output row band ti = wave (rows ti*16 .. ti*16+15).
__device__ __forceinline__ void mm_left_h(const float* __restrict__ src,
                                          float* __restrict__ dst,
                                          int lane, int wave) {
    const int tiBase = wave * 16;
    const int g   = lane >> 4;   // lane group (0/1)
    const int l16 = lane & 15;
    // A = H tile (16x32 per kc); element (M=l16, K) generated in-regs.
    v16h aH0, aH1;
#pragma unroll
    for (int e = 0; e < 16; ++e) {
        const int K = ((e < 8) ? 0 : 16) + (g << 3) + (e & 7);
        aH0[e] = hsign(tiBase + l16, 0  + K);
        aH1[e] = hsign(tiBase + l16, 32 + K);
    }
#pragma unroll
    for (int tj = 0; tj < 4; ++tj) {
        const int tjBase = tj * 16;
        v8f c = {};
#pragma unroll
        for (int kc = 0; kc < 2; ++kc) {
            // B = src tile (32x16): element (K = (e<8?0:16)+g*8+(e&7), N = l16)
            v16h b;
#pragma unroll
            for (int e = 0; e < 16; ++e) {
                const int K = ((e < 8) ? 0 : 16) + (g << 3) + (e & 7);
                b[e] = (_Float16)src[(kc * 32 + K) * LDS_PITCH + tjBase + l16];
            }
            c = __builtin_amdgcn_wmma_f32_16x16x32_f16(
                    false, (kc == 0 ? aH0 : aH1), false, b, (short)0, c, false, false);
        }
#pragma unroll
        for (int v = 0; v < 8; ++v)
            dst[(tiBase + g * 8 + v) * LDS_PITCH + tjBase + l16] = c[v];
    }
}

// dst = src * H64
__device__ __forceinline__ void mm_right_h(const float* __restrict__ src,
                                           float* __restrict__ dst,
                                           int lane, int wave) {
    const int tiBase = wave * 16;
    const int g   = lane >> 4;
    const int l16 = lane & 15;
    // A = src tile (16x32 per kc): element (M=l16, K)
    v16h aT0, aT1;
#pragma unroll
    for (int e = 0; e < 16; ++e) {
        const int K = ((e < 8) ? 0 : 16) + (g << 3) + (e & 7);
        aT0[e] = (_Float16)src[(tiBase + l16) * LDS_PITCH + 0  + K];
        aT1[e] = (_Float16)src[(tiBase + l16) * LDS_PITCH + 32 + K];
    }
#pragma unroll
    for (int tj = 0; tj < 4; ++tj) {
        const int tjBase = tj * 16;
        v8f c = {};
#pragma unroll
        for (int kc = 0; kc < 2; ++kc) {
            // B = H tile (32x16): element (K = (e<8?0:16)+g*8+(e&7), N = l16)
            v16h b;
#pragma unroll
            for (int e = 0; e < 16; ++e) {
                const int K = ((e < 8) ? 0 : 16) + (g << 3) + (e & 7);
                b[e] = hsign(kc * 32 + K, tjBase + l16);
            }
            c = __builtin_amdgcn_wmma_f32_16x16x32_f16(
                    false, (kc == 0 ? aT0 : aT1), false, b, (short)0, c, false, false);
        }
#pragma unroll
        for (int v = 0; v < 8; ++v)
            dst[(tiBase + g * 8 + v) * LDS_PITCH + tjBase + l16] = c[v];
    }
}

// One output pixel (row of the fastfood transform) per workgroup.
// 128 threads = 4 wave32; wave w owns 16-row band w of the 64x64 view.
__global__ __launch_bounds__(128)
void Fastfood_2dConv_kernel(const float* __restrict__ in,    // (16,256,32,32)
                            const float* __restrict__ Bv,    // (4096)
                            const float* __restrict__ Gv,    // (4096)
                            const float* __restrict__ Sv,    // (4096)
                            const float* __restrict__ bias,  // (512)
                            const int*   __restrict__ P,     // (4096)
                            float* __restrict__ out) {       // (16,512,32,32)
    __shared__ float buf0[64 * LDS_PITCH];
    __shared__ float buf1[64 * LDS_PITCH];

    const int tid  = threadIdx.x;
    const int lane = tid & 31;
    const int wave = tid >> 5;

    const int r   = blockIdx.x;      // 0 .. 16383
    const int b   = r >> 10;         // batch
    const int pix = r & 1023;        // ph*32 + pw
    const int ph  = pix >> 5;
    const int pw  = pix & 31;

    // Stage 0: gather im2col patch, scale by B, zero-pad tail. Row layout idx = kk*256 + c.
    for (int idx = tid; idx < 4096; idx += 128) {
        float v = 0.0f;
        if (idx < 2304) {
            const int kk = idx >> 8;        // 0..8 = kh*3+kw
            const int c  = idx & 255;
            const int kh = kk / 3;
            const int kw = kk - kh * 3;
            const int hh = ph + kh - 1;
            const int ww = pw + kw - 1;
            if ((unsigned)hh < 32u && (unsigned)ww < 32u)
                v = Bv[idx] * in[((b * 256 + c) * 32 + hh) * 32 + ww];
        }
        buf0[(idx >> 6) * LDS_PITCH + (idx & 63)] = v;
    }
    __syncthreads();

    // FWHT #1:  Y1 = H * X * H
    mm_left_h (buf0, buf1, lane, wave);   // T1 = H * X
    __syncthreads();
    mm_right_h(buf1, buf0, lane, wave);   // Y1 = T1 * H
    __syncthreads();

    // Permute + G scale:  z[j] = G[j] * Y1[P[j]]
    for (int j = tid; j < 4096; j += 128) {
        const int p = P[j];
        const float z = Gv[j] * buf0[(p >> 6) * LDS_PITCH + (p & 63)];
        buf1[(j >> 6) * LDS_PITCH + (j & 63)] = z;
    }
    __syncthreads();

    // FWHT #2:  Y2 = H * Z * H
    mm_left_h (buf1, buf0, lane, wave);
    __syncthreads();
    mm_right_h(buf0, buf1, lane, wave);
    __syncthreads();

    // Epilogue: out[b][o][ph][pw] = S[o]*Y2[o] + bias[o], o < 512
    for (int o = tid; o < 512; o += 128) {
        out[((b * 512 + o) << 10) + pix] =
            Sv[o] * buf1[(o >> 6) * LDS_PITCH + (o & 63)] + bias[o];
    }
}

extern "C" void kernel_launch(void* const* d_in, const int* in_sizes, int n_in,
                              void* d_out, int out_size, void* d_ws, size_t ws_size,
                              hipStream_t stream) {
    (void)in_sizes; (void)n_in; (void)out_size; (void)d_ws; (void)ws_size;
    const float* in   = (const float*)d_in[0];
    const float* Bv   = (const float*)d_in[1];
    const float* Gv   = (const float*)d_in[2];
    const float* Sv   = (const float*)d_in[3];
    const float* bias = (const float*)d_in[4];
    const int*   P    = (const int*)  d_in[5];
    float* out = (float*)d_out;

    // 16 batches * 32*32 pixels = 16384 rows, one per workgroup.
    Fastfood_2dConv_kernel<<<16384, 128, 0, stream>>>(in, Bv, Gv, Sv, bias, P, out);
}
